// BaseAttention_5360119185484
// MI455X (gfx1250) — matrix operand order repaired and checked by reading, not compiled, not run on hardware
//
#include <hip/hip_runtime.h>

#define DEV __device__ __forceinline__

typedef __attribute__((ext_vector_type(16))) __bf16 v16bf;
typedef __attribute__((ext_vector_type(2)))  __bf16 v2bf;
typedef __attribute__((ext_vector_type(8)))  float  v8f;
typedef int v4i __attribute__((vector_size(16)));

// Fixed problem size (from reference): B=4, N=M=4096, C=256, R=512
constexpr int NQ = 4096;       // query rows per batch
constexpr int MK = 4096;       // key rows per batch
constexpr int CD = 256;        // channel dim (all projection outputs)
constexpr int ROW_SHIFT = 12;  // log2(4096)
constexpr int ROW_MASK  = 4095;

union FragBF {
  unsigned int u[8];
  v16bf v;
};

// native f32 -> bf16 (lowered to v_cvt_pk_bf16_f32)
DEV unsigned short f2bf(float f) {
  return __builtin_bit_cast(unsigned short, (__bf16)f);
}
DEV unsigned int pack2(float lo, float hi) {
  v2bf p;
  p.x = (__bf16)lo;
  p.y = (__bf16)hi;
  return __builtin_bit_cast(unsigned int, p);
}

// 16-bit A-matrix 16x32 — VGPR v, half = lane>>4, K pair base (contiguous, matches B):
DEV int kmapA(int v, int half) { return half * 16 + (v << 1); }
// 16-bit B-matrix 32x16, contiguous k order chosen here (not AMD's table order) — lanes 0-15 hold K=0..15, lanes 16-31 hold K=16..31:
DEV int kmapB(int v, int half) { return half * 16 + (v << 1); }

DEV v8f wmma_bf16(const FragBF& a, const FragBF& b, v8f c) {
  return __builtin_amdgcn_wmma_f32_16x16x32_bf16(
      /*neg_a=*/false, a.v, /*neg_b=*/false, b.v,
      /*c_mod=*/(short)0, c, /*reuse_a=*/false, /*reuse_b=*/false);
}

// ---- gfx1250 async global->LDS copy (ASYNCcnt-tracked), 16B per lane ------
DEV void async_copy16(const void* g, void* l) {
#if __has_builtin(__builtin_amdgcn_global_load_async_to_lds_b128)
  __builtin_amdgcn_global_load_async_to_lds_b128(
      (__attribute__((address_space(1))) v4i*)g,
      (__attribute__((address_space(3))) v4i*)l, 0, 0);
#else
  const unsigned int loff =
      (unsigned int)(__UINTPTR_TYPE__)(__attribute__((address_space(3))) v4i*)l;
  const unsigned long long ga = (unsigned long long)(__UINTPTR_TYPE__)g;
  asm volatile("global_load_async_to_lds_b128 %0, %1, off" ::"v"(loff), "v"(ga)
               : "memory");
#endif
}
DEV void wait_async16() {
#if __has_builtin(__builtin_amdgcn_s_wait_asynccnt)
  __builtin_amdgcn_s_wait_asynccnt(16);
#else
  asm volatile("s_wait_asynccnt 16" ::: "memory");
#endif
}
DEV void wait_async0() {
#if __has_builtin(__builtin_amdgcn_s_wait_asynccnt)
  __builtin_amdgcn_s_wait_asynccnt(0);
#else
  asm volatile("s_wait_asynccnt 0" ::: "memory");
#endif
}

// ---------------------------------------------------------------------------
// Stage 1: OUT[row][o] = sum_k IN[row][k] * W[o][k], fp32 in -> bf16 out.
// Odim fixed = 256. Block = 128 threads (4 waves); each wave computes a
// 16x64 output tile (4 WMMA accumulators) so the A fragment is reused 4x.
// TRANS=true stores OUT as [b][o][m] (for V^T).
// ---------------------------------------------------------------------------
template <int KD, bool TRANS>
__global__ void proj_gemm_f32_bf16(const float* __restrict__ in,
                                   const float* __restrict__ W,
                                   unsigned short* __restrict__ out) {
  const int lane = threadIdx.x & 31;
  const int wave = threadIdx.x >> 5;
  const int half = lane >> 4;
  const int ln   = lane & 15;
  const int rowTile = blockIdx.x * 16;
  const int obase   = wave * 64;  // 4 waves cover Odim = 256

  v8f acc[4] = {};
  for (int kk = 0; kk < KD; kk += 32) {
    FragBF a;
#pragma unroll
    for (int v = 0; v < 8; ++v) {
      const float2 av = *(const float2*)&in[(size_t)(rowTile + ln) * KD + kk + kmapA(v, half)];
      a.u[v] = pack2(av.x, av.y);
    }
#pragma unroll
    for (int t = 0; t < 4; ++t) {
      FragBF b;
#pragma unroll
      for (int v = 0; v < 8; ++v) {
        const float2 bv =
            *(const float2*)&W[(size_t)(obase + t * 16 + ln) * KD + kk + kmapB(v, half)];
        b.u[v] = pack2(bv.x, bv.y);
      }
      acc[t] = wmma_bf16(a, b, acc[t]);
    }
  }
#pragma unroll
  for (int t = 0; t < 4; ++t) {
#pragma unroll
    for (int r = 0; r < 8; ++r) {
      const int grow = rowTile + r + 8 * half;  // row in [B*4096]
      const int col  = obase + t * 16 + ln;
      size_t idx;
      if (TRANS) {
        const int bidx = grow >> ROW_SHIFT;
        const int m    = grow & ROW_MASK;
        idx = ((size_t)bidx * CD + col) * (size_t)MK + m;
      } else {
        idx = (size_t)grow * CD + col;
      }
      out[idx] = f2bf(acc[t][r]);
    }
  }
}

// ---------------------------------------------------------------------------
// Stage 3: bf16 in -> fp32 out GEMM (output projection). 16x64 per wave.
// ---------------------------------------------------------------------------
__global__ void out_gemm_bf16_f32(const unsigned short* __restrict__ in,
                                  const float* __restrict__ W,
                                  float* __restrict__ out) {
  const int lane = threadIdx.x & 31;
  const int wave = threadIdx.x >> 5;
  const int half = lane >> 4;
  const int ln   = lane & 15;
  const int rowTile = blockIdx.x * 16;
  const int obase   = wave * 64;

  v8f acc[4] = {};
  for (int kk = 0; kk < CD; kk += 32) {
    FragBF a;
#pragma unroll
    for (int v = 0; v < 8; ++v)
      a.u[v] = *(const unsigned int*)&in[(size_t)(rowTile + ln) * CD + kk + kmapA(v, half)];
#pragma unroll
    for (int t = 0; t < 4; ++t) {
      FragBF b;
#pragma unroll
      for (int v = 0; v < 8; ++v) {
        const float2 bv =
            *(const float2*)&W[(size_t)(obase + t * 16 + ln) * CD + kk + kmapB(v, half)];
        b.u[v] = pack2(bv.x, bv.y);
      }
      acc[t] = wmma_bf16(a, b, acc[t]);
    }
  }
#pragma unroll
  for (int t = 0; t < 4; ++t)
#pragma unroll
    for (int r = 0; r < 8; ++r)
      out[(size_t)(rowTile + r + 8 * half) * CD + obase + t * 16 + ln] = acc[t][r];
}

// ---------------------------------------------------------------------------
// Stage 2: flash attention with async K/V staging.
// Block = 128 threads (4 waves, 4 consecutive query tiles, same batch).
// Per 32-key step the block stages K (32x256 bf16) and V (256x32 bf16 from
// V^T) into LDS once via global_load_async_to_lds_b128, double-buffered;
// all 4 waves consume the tiles from LDS (4x less L2 traffic).
// ---------------------------------------------------------------------------
constexpr int KROWB  = 512 + 16;        // padded K-tile row bytes (16B aligned)
constexpr int VROWB  = 64 + 16;         // padded V-tile row bytes (16B aligned)
constexpr int KTILEB = 32 * KROWB;      // 16896 B
constexpr int VTILEB = 256 * VROWB;     // 20480 B
constexpr int BUFB   = KTILEB + VTILEB; // 37376 B per buffer

DEV void stage_async(const unsigned short* __restrict__ Km,
                     const unsigned short* __restrict__ Vt,
                     size_t kbase, size_t vbase, int mb, int tid,
                     unsigned char* buf) {
  // K tile: 32 keys x 512B  = 1024 x 16B chunks, 8 per thread
#pragma unroll
  for (int j = 0; j < 8; ++j) {
    const int chunk = j * 128 + tid;
    const int key   = chunk >> 5;          // 32 chunks per key row
    const int coff  = (chunk & 31) << 4;   // byte offset in row
    async_copy16(&Km[kbase + (size_t)(mb + key) * CD + (coff >> 1)],
                 buf + key * KROWB + coff);
  }
  // V tile: 256 channels x 64B = 1024 x 16B chunks, 8 per thread
#pragma unroll
  for (int j = 0; j < 8; ++j) {
    const int chunk = j * 128 + tid;
    const int c     = chunk >> 2;          // 4 chunks per channel row
    const int moff  = (chunk & 3) << 4;    // byte offset in row
    async_copy16(&Vt[vbase + (size_t)c * MK + mb + (moff >> 1)],
                 buf + KTILEB + c * VROWB + moff);
  }
}

__global__ void flash_attn_kernel(const unsigned short* __restrict__ Q,
                                  const unsigned short* __restrict__ Km,
                                  const unsigned short* __restrict__ Vt,
                                  unsigned short* __restrict__ Y,
                                  float scale) {
  const int tid  = threadIdx.x;
  const int lane = tid & 31;
  const int wave = tid >> 5;
  const int half = lane >> 4;
  const int ln   = lane & 15;
  const int grow = (blockIdx.x * 4 + wave) * 16;    // query row in [B*NQ]
  const int b    = (blockIdx.x * 64) >> ROW_SHIFT;  // block-uniform batch

  __shared__ __align__(16) unsigned char smem[2 * BUFB];  // K/V double buffer
  __shared__ unsigned int plds_all[4 * 16 * 16];          // per-wave P transpose
  unsigned int*   plds   = &plds_all[wave * 256];
  unsigned short* plds16 = (unsigned short*)plds;

  // Load Q tile as 8 A-fragments (16 rows x 256 channels), resident in VGPRs
  FragBF qf[8];
#pragma unroll
  for (int f = 0; f < 8; ++f)
#pragma unroll
    for (int v = 0; v < 8; ++v)
      qf[f].u[v] = *(const unsigned int*)&Q[(size_t)(grow + ln) * CD + f * 32 + kmapA(v, half)];

  float mrow[8], lrow[8];
  v8f oacc[16];
  const v8f vzero = {};
#pragma unroll
  for (int r = 0; r < 8; ++r) { mrow[r] = -3.0e38f; lrow[r] = 0.0f; }
#pragma unroll
  for (int t = 0; t < 16; ++t) oacc[t] = vzero;

  const size_t kbase = (size_t)b * MK * CD;
  const size_t vbase = (size_t)b * CD * MK;

  stage_async(Km, Vt, kbase, vbase, 0, tid, smem);  // prologue: buffer 0
  int cur = 0;

  for (int mb = 0; mb < MK; mb += 32) {
    if (mb + 32 < MK) {
      stage_async(Km, Vt, kbase, vbase, mb + 32, tid, smem + (cur ^ 1) * BUFB);
      wait_async16();  // current buffer's 16 older copies have retired
    } else {
      wait_async0();
    }
    __syncthreads();  // current buffer visible to all waves

    const unsigned int* kb = (const unsigned int*)(smem + cur * BUFB);
    const unsigned int* vb = (const unsigned int*)(smem + cur * BUFB + KTILEB);

    // ---- S = Q K^T for 32 keys (two 16x16 tiles), accumulate over C ----
    v8f s0 = vzero, s1 = vzero;
#pragma unroll
    for (int f = 0; f < 8; ++f) {
      FragBF b0, b1;
#pragma unroll
      for (int v = 0; v < 8; ++v) {
        const int cidx = f * 16 + half * 8 + v;       // (f*32 + kmapB)/2 dwords
        b0.u[v] = kb[ln * (KROWB / 4) + cidx];
        b1.u[v] = kb[(16 + ln) * (KROWB / 4) + cidx];
      }
      s0 = wmma_bf16(qf[f], b0, s0);
      s1 = wmma_bf16(qf[f], b1, s1);
    }

    // ---- online softmax over the 32 keys (rows live in 16-lane groups) ----
    float alpha[8];
#pragma unroll
    for (int r = 0; r < 8; ++r) {
      const float v0 = s0[r] * scale;
      const float v1 = s1[r] * scale;
      float mx = fmaxf(v0, v1);
      mx = fmaxf(mx, __shfl_xor(mx, 1));
      mx = fmaxf(mx, __shfl_xor(mx, 2));
      mx = fmaxf(mx, __shfl_xor(mx, 4));
      mx = fmaxf(mx, __shfl_xor(mx, 8));
      const float mnew = fmaxf(mrow[r], mx);
      const float a  = __expf(mrow[r] - mnew);
      const float p0 = __expf(v0 - mnew);
      const float p1 = __expf(v1 - mnew);
      float rs = p0 + p1;
      rs += __shfl_xor(rs, 1);
      rs += __shfl_xor(rs, 2);
      rs += __shfl_xor(rs, 4);
      rs += __shfl_xor(rs, 8);
      lrow[r] = lrow[r] * a + rs;
      mrow[r] = mnew;
      alpha[r] = a;
      // stash P in LDS row-major [row][key] bf16 (D-layout -> memory)
      plds16[(r + 8 * half) * 32 + ln]      = f2bf(p0);
      plds16[(r + 8 * half) * 32 + 16 + ln] = f2bf(p1);
    }
#pragma unroll
    for (int t = 0; t < 16; ++t)
#pragma unroll
      for (int r = 0; r < 8; ++r) oacc[t][r] *= alpha[r];

    // reload P as an A-fragment (16 rows x 32 keys)
    FragBF pa;
#pragma unroll
    for (int v = 0; v < 8; ++v)
      pa.u[v] = plds[ln * 16 + (kmapA(v, half) >> 1)];

    // ---- O += P * V : 16 c-tiles, K-dim = 32 keys, B frags from LDS ----
#pragma unroll
    for (int t = 0; t < 16; ++t) {
      FragBF vf;
#pragma unroll
      for (int v = 0; v < 8; ++v)
        vf.u[v] = vb[(t * 16 + ln) * (VROWB / 4) + half * 8 + v];
      oacc[t] = wmma_bf16(pa, vf, oacc[t]);
    }

    __syncthreads();  // all waves done with buffer before it is restaged
    cur ^= 1;
  }

  // normalize and store Y (bf16)
#pragma unroll
  for (int r = 0; r < 8; ++r) {
    const float inv = 1.0f / lrow[r];
#pragma unroll
    for (int t = 0; t < 16; ++t)
      Y[(size_t)(grow + r + 8 * half) * CD + t * 16 + ln] = f2bf(oacc[t][r] * inv);
  }
}

// ---------------------------------------------------------------------------
extern "C" void kernel_launch(void* const* d_in, const int* in_sizes, int n_in,
                              void* d_out, int out_size, void* d_ws, size_t ws_size,
                              hipStream_t stream) {
  (void)in_sizes; (void)n_in; (void)out_size; (void)ws_size;
  const int B = 4;
  const float scale = 0.0625f;  // 256^-0.5

  const float* x   = (const float*)d_in[0];
  const float* ref = (const float*)d_in[1];
  const float* Wq  = (const float*)d_in[2];
  const float* Wk  = (const float*)d_in[3];
  const float* Wv  = (const float*)d_in[4];
  const float* Wo  = (const float*)d_in[5];
  float* out = (float*)d_out;

  unsigned short* Qb = (unsigned short*)d_ws;               // [B*NQ*CD] bf16
  unsigned short* Kb = Qb + (size_t)B * NQ * CD;            // [B*MK*CD]
  unsigned short* Vb = Kb + (size_t)B * MK * CD;            // [B*CD*MK] (V^T)
  unsigned short* Yb = Vb + (size_t)B * CD * MK;            // [B*NQ*CD]

  const dim3 blk(128);  // 4 waves
  // Q = x @ Wq^T   (K=256)
  proj_gemm_f32_bf16<256, false><<<dim3(B * NQ / 16), blk, 0, stream>>>(x, Wq, Qb);
  // K = ref @ Wk^T (K=512)
  proj_gemm_f32_bf16<512, false><<<dim3(B * MK / 16), blk, 0, stream>>>(ref, Wk, Kb);
  // V^T = (ref @ Wv^T)^T
  proj_gemm_f32_bf16<512, true><<<dim3(B * MK / 16), blk, 0, stream>>>(ref, Wv, Vb);
  // flash attention: B*NQ/16 = 1024 query tiles, 4 per block
  flash_attn_kernel<<<dim3(B * NQ / 16 / 4), blk, 0, stream>>>(Qb, Kb, Vb, Yb, scale);
  // out = Y @ Wo^T
  out_gemm_bf16_f32<<<dim3(B * NQ / 16), blk, 0, stream>>>(Yb, Wo, out);
}
